// Model_34110630264965
// MI455X (gfx1250) — matrix operand order repaired and checked by reading, not compiled, not run on hardware
//
#include <hip/hip_runtime.h>
#include <stdint.h>

// ---------------------------------------------------------------------------
// MoE block for MI455X (gfx1250, wave32): WMMA bf16 16x16x32 GEMMs with
// double-buffered LDS weight tiles staged by the Tensor Data Mover (TDM)
// when available, else gfx1250 async-to-LDS DMA, else plain staging.
// ---------------------------------------------------------------------------

typedef __attribute__((ext_vector_type(16))) __bf16 v16bf;
typedef __attribute__((ext_vector_type(8)))  float  v8f;

union Frag16 {
  v16bf v;
  uint4 q[2];
};

static __device__ __forceinline__ unsigned short f2bf(float f) {
  unsigned int u = __float_as_uint(f);
  u = (u + 0x7FFFu + ((u >> 16) & 1u)) >> 16;   // round-to-nearest-even
  return (unsigned short)u;
}

#define T_TOK 2048
#define DIM   1024
#define FEXP  2048
#define FSH   4096
#define NEXP  8

#define BK    32      // K per stage
#define BN    64      // N per block
#define BM    128     // M per block (8 waves x 16)
#define LSTR  72      // padded LDS row stride in bf16 (64 data + 8 pad)

// ---------------- staging path selection (guarded, with safe fallbacks) -----
#if defined(__has_builtin)
#if __has_builtin(__builtin_amdgcn_tensor_load_to_lds) && \
    __has_builtin(__builtin_amdgcn_s_wait_tensorcnt)
#define USE_TDM 1
#endif
#if __has_builtin(__builtin_amdgcn_global_load_async_to_lds_b128) && \
    __has_builtin(__builtin_amdgcn_s_wait_asynccnt)
#define USE_ASYNC_LDS 1
#endif
#endif
#ifndef USE_TDM
#define USE_TDM 0
#endif
#ifndef USE_ASYNC_LDS
#define USE_ASYNC_LDS 0
#endif

#if USE_ASYNC_LDS
typedef int v4i_vs __attribute__((vector_size(16)));
typedef __attribute__((address_space(1))) v4i_vs* g_v4i_ptr;   // global
typedef __attribute__((address_space(3))) v4i_vs* l_v4i_ptr;   // LDS
#endif

static __device__ __forceinline__ void copy16_g2l(
    const unsigned short* gp, unsigned short* lp) {
#if USE_ASYNC_LDS
  __builtin_amdgcn_global_load_async_to_lds_b128(
      (g_v4i_ptr)gp, (l_v4i_ptr)lp, 0, 0);
#else
  *(uint4*)lp = *(const uint4*)gp;
#endif
}

#if USE_TDM
typedef unsigned int u32x4 __attribute__((vector_size(16)));
typedef int i32x8 __attribute__((vector_size(32)));
typedef int i32x4 __attribute__((vector_size(16)));

// Stage a 32(K) x 64(N) bf16 tile (row stride F elements) into LDS with the
// TDM; pad 4 dwords after every 32 dwords -> LSTR=72 bf16 LDS row stride.
// D# bitfields per CDNA5 ISA ch.8 (group0 128b, group1 256b).
static __device__ __forceinline__ void tdm_stage(
    const unsigned short* gp, const unsigned short* lp, int F) {
  unsigned long long ga =
      (unsigned long long)(size_t)gp;                  // tile global address
  unsigned int la = (unsigned int)(size_t)(
      __attribute__((address_space(3))) const unsigned short*)lp;  // LDS byte addr

  u32x4 g0;
  g0[0] = 1u;                                          // count=1, user mode
  g0[1] = la;                                          // lds_addr
  g0[2] = (unsigned int)ga;                            // global_addr[31:0]
  g0[3] = (unsigned int)(ga >> 32) | (2u << 30);       // addr[56:32] | type=2

  const unsigned int td0 = (unsigned int)F;            // tensor_dim0
  const unsigned int td1 = 0xFFFFu;                    // tensor_dim1 (no OOB)
  const unsigned long long s0 = (unsigned long long)F; // dim0 stride (elems)

  i32x8 g1;
  g1[0] = (int)((1u << 16)      // data_size = 2 bytes
              | (1u << 20)      // pad_enable
              | (4u << 22)      // pad_interval: 32 dwords
              | (3u << 25));    // pad_amount: 4 dwords
  g1[1] = (int)((td0 & 0xFFFFu) << 16);                      // td0 lo16
  g1[2] = (int)((td0 >> 16) | ((td1 & 0xFFFFu) << 16));      // td0 hi | td1 lo
  g1[3] = (int)((td1 >> 16) | (64u << 16));                  // td1 hi | tile_dim0=64
  g1[4] = (int)(32u);                                        // tile_dim1=32
  g1[5] = (int)(unsigned int)s0;                             // dim0 stride lo32
  g1[6] = (int)((unsigned int)(s0 >> 32) & 0xFFFFu);         // stride hi16
  g1[7] = 0;
  i32x4 gz4 = {0, 0, 0, 0};
  i32x8 gz8 = {0, 0, 0, 0, 0, 0, 0, 0};
  __builtin_amdgcn_tensor_load_to_lds(g0, g1, gz4, gz4, gz8, 0);
}
#endif

static __device__ __forceinline__ void wait_stage(int wave) {
#if USE_TDM
  if (wave == 0) __builtin_amdgcn_s_wait_tensorcnt(0);
#elif USE_ASYNC_LDS
  __builtin_amdgcn_s_wait_asynccnt(0);
#endif
  __syncthreads();
}

// ---------------------------------------------------------------- conversion
__global__ void cvt_bf16_kernel(const float* __restrict__ src,
                                unsigned short* __restrict__ dst, int n4) {
  int i = blockIdx.x * blockDim.x + threadIdx.x;
  if (i < n4) {
    float4 v = ((const float4*)src)[i];
    uint2 p;
    p.x = (unsigned)f2bf(v.x) | ((unsigned)f2bf(v.y) << 16);
    p.y = (unsigned)f2bf(v.z) | ((unsigned)f2bf(v.w) << 16);
    ((uint2*)dst)[i] = p;
  }
}

__global__ void zero_kernel(float* __restrict__ p, int n4) {
  int i = blockIdx.x * blockDim.x + threadIdx.x;
  if (i < n4) ((float4*)p)[i] = make_float4(0.f, 0.f, 0.f, 0.f);
}

// ---------------------------------------------------------------- router
__global__ void __launch_bounds__(256) router_kernel(
    const float* __restrict__ x, const float* __restrict__ gw,
    const float* __restrict__ segw, float* __restrict__ logits,
    float* __restrict__ comb, float* __restrict__ shg) {
  const int lane = threadIdx.x & 31;
  const int wave = threadIdx.x >> 5;
  const int t = blockIdx.x * 8 + wave;

  float acc[NEXP] = {0.f, 0.f, 0.f, 0.f, 0.f, 0.f, 0.f, 0.f};
  float sacc = 0.f;
  for (int d = lane; d < DIM; d += 32) {
    float xv = x[(size_t)t * DIM + d];
    const float4* g4 = (const float4*)(gw + (size_t)d * NEXP);
    float4 a = g4[0], b = g4[1];
    acc[0] += xv * a.x; acc[1] += xv * a.y; acc[2] += xv * a.z; acc[3] += xv * a.w;
    acc[4] += xv * b.x; acc[5] += xv * b.y; acc[6] += xv * b.z; acc[7] += xv * b.w;
    sacc += xv * segw[d];
  }
#pragma unroll
  for (int off = 16; off >= 1; off >>= 1) {
#pragma unroll
    for (int e = 0; e < NEXP; ++e) acc[e] += __shfl_xor(acc[e], off, 32);
    sacc += __shfl_xor(sacc, off, 32);
  }
  if (lane == 0) {
#pragma unroll
    for (int e = 0; e < NEXP; ++e) logits[(size_t)t * NEXP + e] = acc[e];
    int i1 = 0;
#pragma unroll
    for (int e = 1; e < NEXP; ++e) if (acc[e] > acc[i1]) i1 = e;
    int i2 = (i1 == 0) ? 1 : 0;
#pragma unroll
    for (int e = 0; e < NEXP; ++e) if (e != i1 && acc[e] > acc[i2]) i2 = e;
    const float e2 = __expf(acc[i2] - acc[i1]);
    const float w1 = 1.f / (1.f + e2);
    const float w2 = e2 / (1.f + e2);
#pragma unroll
    for (int e = 0; e < NEXP; ++e)
      comb[(size_t)t * NEXP + e] = (e == i1) ? w1 : ((e == i2) ? w2 : 0.f);
    shg[t] = 1.f / (1.f + __expf(-sacc));
  }
}

// ------------------------------------------------------- gate/up fused GEMM
__global__ void __launch_bounds__(256) gateup_kernel(
    const unsigned short* __restrict__ xb,   // [T, DIM] bf16
    const unsigned short* __restrict__ wg,   // [DIM, F] bf16
    const unsigned short* __restrict__ wu,   // [DIM, F] bf16
    unsigned short* __restrict__ h,          // [T, F] bf16
    int F) {
  __shared__ __align__(16) unsigned short lg[2][BK * LSTR];
  __shared__ __align__(16) unsigned short lu[2][BK * LSTR];

  const int tid  = threadIdx.x;
  const int lane = tid & 31;
  const int wave = tid >> 5;
  const int n0 = blockIdx.x * BN;
  const int t0 = blockIdx.y * BM + wave * 16;

  const int srow   = tid >> 3;          // 0..31 (K row)
  const int schunk = (tid & 7) * 8;     // bf16 element offset within row
  const int loff   = srow * LSTR + schunk;
  (void)loff;

  v8f gacc[4] = {};
  v8f uacc[4] = {};

  const unsigned short* ap0 =
      xb + (size_t)(t0 + (lane & 15)) * DIM + ((lane >> 4) << 3);

  auto stage = [&](int buf, int k) {
#if USE_TDM
    if (wave == 0) {
      tdm_stage(wg + (size_t)k * F + n0, &lg[buf][0], F);
      tdm_stage(wu + (size_t)k * F + n0, &lu[buf][0], F);
    }
#else
    const size_t gofs = (size_t)(k + srow) * F + n0 + schunk;
    copy16_g2l(wg + gofs, &lg[buf][loff]);
    copy16_g2l(wu + gofs, &lu[buf][loff]);
#endif
  };

  stage(0, 0);   // prologue

  for (int k0 = 0; k0 < DIM; k0 += BK) {
    const int cur = (k0 >> 5) & 1;
    wait_stage(wave);                    // buffer `cur` ready, prev reads done
    if (k0 + BK < DIM) stage(cur ^ 1, k0 + BK);

    Frag16 a;
    a.q[0] = *(const uint4*)(ap0 + k0);
    a.q[1] = *(const uint4*)(ap0 + k0 + 16);
    const unsigned short* bg0 = &lg[cur][((lane >> 4) << 3) * LSTR + (lane & 15)];
    const unsigned short* bu0 = &lu[cur][((lane >> 4) << 3) * LSTR + (lane & 15)];

    Frag16 bg[4], bu[4];
#pragma unroll
    for (int i = 0; i < 4; ++i) {        // batch all LDS frag loads first
#pragma unroll
      for (int s = 0; s < 16; ++s) {
        const int ko = (s < 8 ? s : s + 8) * LSTR + i * 16;
        ((unsigned short*)&bg[i])[s] = bg0[ko];
        ((unsigned short*)&bu[i])[s] = bu0[ko];
      }
    }
    asm volatile("" ::: "memory");       // keep all loads ahead of the WMMAs
#pragma unroll
    for (int i = 0; i < 4; ++i) {        // then back-to-back WMMAs
      gacc[i] = __builtin_amdgcn_wmma_f32_16x16x32_bf16(
          false, a.v, false, bg[i].v, (short)0, gacc[i], false, false);
      uacc[i] = __builtin_amdgcn_wmma_f32_16x16x32_bf16(
          false, a.v, false, bu[i].v, (short)0, uacc[i], false, false);
    }
  }

  const int col = n0 + (lane & 15);
  const int rbase = t0 + ((lane >> 4) << 3);
#pragma unroll
  for (int i = 0; i < 4; ++i) {
#pragma unroll
    for (int r = 0; r < 8; ++r) {
      float g = gacc[i][r], u = uacc[i][r];
      float hv = (g / (1.f + __expf(-g))) * u;   // silu(g) * u
      h[(size_t)(rbase + r) * F + col + i * 16] = f2bf(hv);
    }
  }
}

// ------------------------------------------------------------- down GEMM
__global__ void __launch_bounds__(256) down_kernel(
    const unsigned short* __restrict__ hb,   // [T, K] bf16
    const unsigned short* __restrict__ wd,   // [K, DIM] bf16
    float* __restrict__ out,                 // [T, DIM] f32 accum
    const float* __restrict__ scale, int sstride, int K) {
  __shared__ __align__(16) unsigned short lb[2][BK * LSTR];

  const int tid  = threadIdx.x;
  const int lane = tid & 31;
  const int wave = tid >> 5;
  const int n0 = blockIdx.x * BN;
  const int t0 = blockIdx.y * BM + wave * 16;

  const int srow   = tid >> 3;
  const int schunk = (tid & 7) * 8;
  const int loff   = srow * LSTR + schunk;
  (void)loff;

  v8f acc[4] = {};
  const unsigned short* ap0 =
      hb + (size_t)(t0 + (lane & 15)) * K + ((lane >> 4) << 3);

  auto stage = [&](int buf, int k) {
#if USE_TDM
    if (wave == 0) tdm_stage(wd + (size_t)k * DIM + n0, &lb[buf][0], DIM);
#else
    const size_t gofs = (size_t)(k + srow) * DIM + n0 + schunk;
    copy16_g2l(wd + gofs, &lb[buf][loff]);
#endif
  };

  stage(0, 0);

  for (int k0 = 0; k0 < K; k0 += BK) {
    const int cur = (k0 >> 5) & 1;
    wait_stage(wave);
    if (k0 + BK < K) stage(cur ^ 1, k0 + BK);

    Frag16 a;
    a.q[0] = *(const uint4*)(ap0 + k0);
    a.q[1] = *(const uint4*)(ap0 + k0 + 16);
    const unsigned short* b0 = &lb[cur][((lane >> 4) << 3) * LSTR + (lane & 15)];

    Frag16 b[4];
#pragma unroll
    for (int i = 0; i < 4; ++i) {
#pragma unroll
      for (int s = 0; s < 16; ++s)
        ((unsigned short*)&b[i])[s] = b0[(s < 8 ? s : s + 8) * LSTR + i * 16];
    }
    asm volatile("" ::: "memory");       // keep all loads ahead of the WMMAs
#pragma unroll
    for (int i = 0; i < 4; ++i) {
      acc[i] = __builtin_amdgcn_wmma_f32_16x16x32_bf16(
          false, a.v, false, b[i].v, (short)0, acc[i], false, false);
    }
  }

  const int col = n0 + (lane & 15);
  const int rbase = t0 + ((lane >> 4) << 3);
#pragma unroll
  for (int r = 0; r < 8; ++r) {
    const int row = rbase + r;
    const float s = scale[(size_t)row * sstride];
#pragma unroll
    for (int i = 0; i < 4; ++i)
      out[(size_t)row * DIM + col + i * 16] += s * acc[i][r];
  }
}

// ---------------------------------------------------------------------------
extern "C" void kernel_launch(void* const* d_in, const int* in_sizes, int n_in,
                              void* d_out, int out_size, void* d_ws,
                              size_t ws_size, hipStream_t stream) {
  const float* x    = (const float*)d_in[0];   // [2,1024,1024]
  const float* gw   = (const float*)d_in[1];   // [1024, 8]
  const float* egw  = (const float*)d_in[2];   // [8, 1024, 2048]
  const float* euw  = (const float*)d_in[3];   // [8, 1024, 2048]
  const float* edw  = (const float*)d_in[4];   // [8, 2048, 1024]
  const float* sgw  = (const float*)d_in[5];   // [1024, 4096]
  const float* suw  = (const float*)d_in[6];   // [1024, 4096]
  const float* sdw  = (const float*)d_in[7];   // [4096, 1024]
  const float* segw = (const float*)d_in[8];   // [1024, 1]

  float* out    = (float*)d_out;               // [T, DIM]
  float* logits = out + (size_t)T_TOK * DIM;   // [T, 8]

  char*  ws  = (char*)d_ws;
  size_t off = 0;
  auto alloc = [&](size_t bytes) -> char* {
    char* p = ws + off;
    off += (bytes + 255) & ~(size_t)255;
    return p;
  };
  unsigned short* xb  = (unsigned short*)alloc((size_t)T_TOK * DIM * 2);
  unsigned short* egb = (unsigned short*)alloc((size_t)NEXP * DIM * FEXP * 2);
  unsigned short* eub = (unsigned short*)alloc((size_t)NEXP * DIM * FEXP * 2);
  unsigned short* edb = (unsigned short*)alloc((size_t)NEXP * FEXP * DIM * 2);
  unsigned short* sgb = (unsigned short*)alloc((size_t)DIM * FSH * 2);
  unsigned short* sub = (unsigned short*)alloc((size_t)DIM * FSH * 2);
  unsigned short* sdb = (unsigned short*)alloc((size_t)FSH * DIM * 2);
  unsigned short* hb  = (unsigned short*)alloc((size_t)T_TOK * FSH * 2);
  float* comb = (float*)alloc((size_t)T_TOK * NEXP * 4);
  float* shg  = (float*)alloc((size_t)T_TOK * 4);

  auto cvt = [&](const float* s, unsigned short* d, size_t n) {
    int n4 = (int)(n / 4);
    cvt_bf16_kernel<<<(n4 + 255) / 256, 256, 0, stream>>>(s, d, n4);
  };
  cvt(x,   xb,  (size_t)T_TOK * DIM);
  cvt(egw, egb, (size_t)NEXP * DIM * FEXP);
  cvt(euw, eub, (size_t)NEXP * DIM * FEXP);
  cvt(edw, edb, (size_t)NEXP * FEXP * DIM);
  cvt(sgw, sgb, (size_t)DIM * FSH);
  cvt(suw, sub, (size_t)DIM * FSH);
  cvt(sdw, sdb, (size_t)FSH * DIM);

  router_kernel<<<T_TOK / 8, 256, 0, stream>>>(x, gw, segw, logits, comb, shg);

  {
    int n4 = T_TOK * DIM / 4;
    zero_kernel<<<(n4 + 255) / 256, 256, 0, stream>>>(out, n4);
  }

  for (int e = 0; e < NEXP; ++e) {
    const unsigned short* wgp = egb + (size_t)e * DIM * FEXP;
    const unsigned short* wup = eub + (size_t)e * DIM * FEXP;
    const unsigned short* wdp = edb + (size_t)e * FEXP * DIM;
    gateup_kernel<<<dim3(FEXP / BN, T_TOK / BM), 256, 0, stream>>>(
        xb, wgp, wup, hb, FEXP);
    down_kernel<<<dim3(DIM / BN, T_TOK / BM), 256, 0, stream>>>(
        hb, wdp, out, comb + e, NEXP, FEXP);
  }

  gateup_kernel<<<dim3(FSH / BN, T_TOK / BM), 256, 0, stream>>>(
      xb, sgb, sub, hb, FSH);
  down_kernel<<<dim3(DIM / BN, T_TOK / BM), 256, 0, stream>>>(
      hb, sdb, out, shg, 1, FSH);
}
